// Packer_29317446762977
// MI455X (gfx1250) — matrix operand order repaired and checked by reading, not compiled, not run on hardware
//
#include <hip/hip_runtime.h>
#include <hip/hip_bf16.h>
#include <stdint.h>
#include <stddef.h>

// ---- problem constants (from reference) ----
#define HD    128
#define LLEN  2048
#define BATCH 4
#define KNBR  30
#define BL    (BATCH * LLEN)      // 8192 nodes
#define SCALE_INV (1.0f / 30.0f)

typedef __attribute__((ext_vector_type(16))) __bf16  v16bf;
typedef __attribute__((ext_vector_type(8)))  float   v8f;
typedef __attribute__((ext_vector_type(8)))  uint32_t v8u;

// ---- packed-weight workspace layout (uint32 units) ----
// All weights stored fragment-major: dst[(((kc*NT)+nt)*32 + lane)*8 + v]
enum : int {
  OFF_W1   = 0,
  OFF_W2   = OFF_W1  + 192 * 128,   // W1: K=384 (KC=12), N=128 (NT=8)
  OFF_W3   = OFF_W2  +  64 * 128,
  OFF_W11  = OFF_W3  +  64 * 128,
  OFF_W12  = OFF_W11 + 192 * 128,
  OFF_W13  = OFF_W12 +  64 * 128,
  OFF_WIN  = OFF_W13 +  64 * 128,   // Win: KC=4, NT=32
  OFF_WOUT = OFF_WIN +  64 * 512,   // Wout: KC=16, NT=8
  OFF_END  = OFF_WOUT + 256 * 128,
};

// ---------------- helpers ----------------
__device__ __forceinline__ uint32_t f32_to_bf16_bits(float f) {
  uint32_t u = __builtin_bit_cast(uint32_t, f);
  uint32_t r = u + 0x7FFFu + ((u >> 16) & 1u);   // round-to-nearest-even
  return r >> 16;
}
__device__ __forceinline__ uint32_t pack_bf16x2(float lo, float hi) {
  return f32_to_bf16_bits(lo) | (f32_to_bf16_bits(hi) << 16);
}
__device__ __forceinline__ float gelu_erf(float x) {
  return 0.5f * x * (1.0f + erff(x * 0.70710678118654752f));
}

// A-fragment (16x32 bf16) from an LDS bf16 matrix stored row-major
// (strideH halfwords per row), per ISA 05_wmma 16-bit A layout:
// lane: m = lane&15, half = lane>>4; VGPR v holds K = kc*32 + ((v&4)<<2) + half*8 + (v&3)*2, +1
__device__ __forceinline__ v16bf load_a_frag(const uint32_t* ldsW, int strideH,
                                             int mTile, int kc, int lane) {
  const int m = mTile * 16 + (lane & 15);
  const int half = lane >> 4;
  v8u w;
#pragma unroll
  for (int v = 0; v < 8; ++v) {
    const int k0 = kc * 32 + ((v & 4) << 2) + half * 8 + (v & 3) * 2;
    w[v] = ldsW[(m * strideH + k0) >> 1];
  }
  return __builtin_bit_cast(v16bf, w);
}

// B-fragment (32x16 bf16) from fragment-major packed weights: each lane reads
// 8 contiguous dwords (two global_load_b128); a wave reads one 1KB block.
__device__ __forceinline__ v16bf load_b_frag(const uint32_t* __restrict__ Wp, int NT,
                                             int nTile, int kc, int lane) {
  const uint32_t* p = Wp + ((((size_t)kc * NT + nTile) * 32 + lane) << 3);
  const uint4 lo = *(const uint4*)(p);
  const uint4 hi = *(const uint4*)(p + 4);
  v8u w;
  w[0] = lo.x; w[1] = lo.y; w[2] = lo.z; w[3] = lo.w;
  w[4] = hi.x; w[5] = hi.y; w[6] = hi.z; w[7] = hi.w;
  return __builtin_bit_cast(v16bf, w);
}

__device__ __forceinline__ v8f wmma_bf16(v16bf a, v16bf b, v8f c) {
  return __builtin_amdgcn_wmma_f32_16x16x32_bf16(false, a, false, b, (short)0, c,
                                                 false, false);
}

// D-tile (16x16 f32) -> bias + GELU -> bf16 into LDS (strideH halfwords/row).
// C/D layout: VGPR i => row = mTile*16 + (lane>>4)*8 + i, col = nTile*16 + (lane&15)
__device__ __forceinline__ void store_gelu_tile(uint32_t* yW, int strideH, v8f acc,
                                                int mTile, int nTile,
                                                const float* __restrict__ bias, int lane) {
  __bf16* y = (__bf16*)yW;
  const int col = nTile * 16 + (lane & 15);
  const float bv = bias[col];
#pragma unroll
  for (int i = 0; i < 8; ++i) {
    const int row = mTile * 16 + (lane >> 4) * 8 + i;
    const uint32_t bits = f32_to_bf16_bits(gelu_erf(acc[i] + bv));
    y[row * strideH + col] = __builtin_bit_cast(__bf16, (unsigned short)bits);
  }
}

// ---- stage x = concat(center, h_E, gathered) as 32x384 bf16 in LDS ----
__device__ __forceinline__ void stage_x(uint32_t* xW, const float* __restrict__ feat,
                                        const float* __restrict__ h_E,
                                        const int* __restrict__ E_idx,
                                        int node, int b, int tid, int* idxSh) {
  for (int i = tid; i < 2 * 192; i += 128) xW[30 * 192 + i] = 0;  // zero pad rows 30,31
  if (tid < 32) idxSh[tid] = (tid < KNBR) ? E_idx[(size_t)node * KNBR + tid] : 0;
  __syncthreads();
  const float* hvc = feat + (size_t)node * HD;
  for (int i = tid; i < KNBR * 32; i += 128) {
    const int r = i >> 5, q = i & 31;
    const float4 f = *(const float4*)(hvc + q * 4);
    xW[r * 192 + q * 2]     = pack_bf16x2(f.x, f.y);
    xW[r * 192 + q * 2 + 1] = pack_bf16x2(f.z, f.w);
    const float4 e = *(const float4*)(h_E + ((size_t)node * KNBR + r) * HD + q * 4);
    xW[r * 192 + 64 + q * 2]     = pack_bf16x2(e.x, e.y);
    xW[r * 192 + 64 + q * 2 + 1] = pack_bf16x2(e.z, e.w);
    const float4 g = *(const float4*)(feat + ((size_t)(b * LLEN + idxSh[r])) * HD + q * 4);
    xW[r * 192 + 128 + q * 2]     = pack_bf16x2(g.x, g.y);
    xW[r * 192 + 128 + q * 2 + 1] = pack_bf16x2(g.z, g.w);
  }
  __syncthreads();
}

// ---- 3-layer edge MLP over a 32-row tile: x[32x384] -> m3[32x128] (f32, +bias) ----
__device__ __forceinline__ void edge_mlp_32(
    const uint32_t* xW, uint32_t* y1W, uint32_t* y2W, float* m3,
    const uint32_t* __restrict__ W1p, const float* __restrict__ b1,
    const uint32_t* __restrict__ W2p, const float* __restrict__ b2,
    const uint32_t* __restrict__ W3p, const float* __restrict__ b3, int tid) {
  const int lane = tid & 31, wv = tid >> 5;
  // layer 1: [32x384]x[384x128]
  {
    v8f acc[2][2] = {};
#pragma unroll
    for (int kc = 0; kc < 12; ++kc) {
      const v16bf a0 = load_a_frag(xW, 384, 0, kc, lane);
      const v16bf a1 = load_a_frag(xW, 384, 1, kc, lane);
      const v16bf w0 = load_b_frag(W1p, 8, 2 * wv + 0, kc, lane);
      const v16bf w1 = load_b_frag(W1p, 8, 2 * wv + 1, kc, lane);
      acc[0][0] = wmma_bf16(a0, w0, acc[0][0]);
      acc[0][1] = wmma_bf16(a0, w1, acc[0][1]);
      acc[1][0] = wmma_bf16(a1, w0, acc[1][0]);
      acc[1][1] = wmma_bf16(a1, w1, acc[1][1]);
    }
#pragma unroll
    for (int mt = 0; mt < 2; ++mt)
#pragma unroll
      for (int t = 0; t < 2; ++t)
        store_gelu_tile(y1W, 128, acc[mt][t], mt, 2 * wv + t, b1, lane);
  }
  __syncthreads();
  // layer 2: [32x128]x[128x128]
  {
    v8f acc[2][2] = {};
#pragma unroll
    for (int kc = 0; kc < 4; ++kc) {
      const v16bf a0 = load_a_frag(y1W, 128, 0, kc, lane);
      const v16bf a1 = load_a_frag(y1W, 128, 1, kc, lane);
      const v16bf w0 = load_b_frag(W2p, 8, 2 * wv + 0, kc, lane);
      const v16bf w1 = load_b_frag(W2p, 8, 2 * wv + 1, kc, lane);
      acc[0][0] = wmma_bf16(a0, w0, acc[0][0]);
      acc[0][1] = wmma_bf16(a0, w1, acc[0][1]);
      acc[1][0] = wmma_bf16(a1, w0, acc[1][0]);
      acc[1][1] = wmma_bf16(a1, w1, acc[1][1]);
    }
#pragma unroll
    for (int mt = 0; mt < 2; ++mt)
#pragma unroll
      for (int t = 0; t < 2; ++t)
        store_gelu_tile(y2W, 128, acc[mt][t], mt, 2 * wv + t, b2, lane);
  }
  __syncthreads();
  // layer 3: [32x128]x[128x128] -> f32 (no activation)
  {
    v8f acc[2][2] = {};
#pragma unroll
    for (int kc = 0; kc < 4; ++kc) {
      const v16bf a0 = load_a_frag(y2W, 128, 0, kc, lane);
      const v16bf a1 = load_a_frag(y2W, 128, 1, kc, lane);
      const v16bf w0 = load_b_frag(W3p, 8, 2 * wv + 0, kc, lane);
      const v16bf w1 = load_b_frag(W3p, 8, 2 * wv + 1, kc, lane);
      acc[0][0] = wmma_bf16(a0, w0, acc[0][0]);
      acc[0][1] = wmma_bf16(a0, w1, acc[0][1]);
      acc[1][0] = wmma_bf16(a1, w0, acc[1][0]);
      acc[1][1] = wmma_bf16(a1, w1, acc[1][1]);
    }
#pragma unroll
    for (int mt = 0; mt < 2; ++mt)
#pragma unroll
      for (int t = 0; t < 2; ++t) {
        const int nt = 2 * wv + t;
        const int col = nt * 16 + (lane & 15);
        const float bv = b3[col];
#pragma unroll
        for (int i = 0; i < 8; ++i) {
          const int row = mt * 16 + (lane >> 4) * 8 + i;
          m3[row * HD + col] = acc[mt][t][i] + bv;
        }
      }
  }
  __syncthreads();
}

// -------- kernel 1: weight packing (f32 -> fragment-major paired bf16 dwords) --------
// dst[(((kc*NT)+nt)*32 + lane)*8 + v] = pack(W[k][n], W[k+1][n])
//   with n = nt*16 + (lane&15), k = kc*32 + ((v&4)<<2) + (lane>>4)*8 + (v&3)*2
__global__ __launch_bounds__(256) void pack_w_kernel(const float* __restrict__ src,
                                                     uint32_t* __restrict__ dst,
                                                     int KC, int NT, int N) {
  const int p = blockIdx.x * 256 + threadIdx.x;
  if (p >= KC * NT * 256) return;
  const int v    = p & 7;
  const int lane = (p >> 3) & 31;
  const int frag = p >> 8;
  const int nt = frag % NT, kc = frag / NT;
  const int n = nt * 16 + (lane & 15);
  const int k = kc * 32 + ((v & 4) << 2) + (lane >> 4) * 8 + (v & 3) * 2;
  dst[p] = pack_bf16x2(src[(size_t)k * N + n], src[(size_t)(k + 1) * N + n]);
}

// ---------------- kernel 2: node message + LN1 ----------------
__global__ __launch_bounds__(128) void enc_node_kernel(
    const float* __restrict__ h_V, const float* __restrict__ h_E,
    const int* __restrict__ E_idx, const float* __restrict__ mask_attend,
    const uint32_t* __restrict__ W1p, const float* __restrict__ b1,
    const uint32_t* __restrict__ W2p, const float* __restrict__ b2,
    const uint32_t* __restrict__ W3p, const float* __restrict__ b3,
    const float* __restrict__ g1, const float* __restrict__ be1,
    float* __restrict__ hV1) {
  __shared__ uint32_t xW[32 * 192];
  __shared__ uint32_t y1W[32 * 64];
  __shared__ uint32_t y2W[32 * 64];
  __shared__ float m3[32 * HD];
  __shared__ float red[HD], red2[HD];
  __shared__ int idxSh[32];

  const int tid = threadIdx.x;
  const int node = blockIdx.x;
  const int b = node >> 11;  // L = 2048
  __builtin_prefetch(W1p, 0, 0);

  stage_x(xW, h_V, h_E, E_idx, node, b, tid, idxSh);
  edge_mlp_32(xW, y1W, y2W, m3, W1p, b1, W2p, b2, W3p, b3, tid);

  // masked sum over neighbors, residual, LayerNorm (per node, 128 dims)
  const int n = tid;
  float s = 0.0f;
  for (int r = 0; r < KNBR; ++r) s += m3[r * HD + n] * mask_attend[(size_t)node * KNBR + r];
  const float* hvc = h_V + (size_t)node * HD;
  const float u = hvc[n] + s * SCALE_INV;
  red[n] = u; red2[n] = u * u;
  __syncthreads();
#pragma unroll
  for (int st = 64; st > 0; st >>= 1) {
    if (n < st) { red[n] += red[n + st]; red2[n] += red2[n + st]; }
    __syncthreads();
  }
  const float mean = red[0] * (1.0f / 128.0f);
  const float var  = red2[0] * (1.0f / 128.0f) - mean * mean;
  hV1[(size_t)node * HD + n] = (u - mean) * rsqrtf(var + 1e-5f) * g1[n] + be1[n];
}

// ---------------- kernel 3: position-wise FFN + LN2 + mask_V ----------------
__global__ __launch_bounds__(128) void enc_ffn_kernel(
    const float* __restrict__ hV1,
    const uint32_t* __restrict__ Winp, const float* __restrict__ bin,
    const uint32_t* __restrict__ Woutp, const float* __restrict__ bout,
    const float* __restrict__ g2, const float* __restrict__ be2,
    const float* __restrict__ mask_V,
    float* __restrict__ hV2, float* __restrict__ out_hV) {
  __shared__ uint32_t xW[16 * 64];     // 16x128 bf16
  __shared__ float xF[16 * HD];        // f32 residual copy
  __shared__ uint32_t t1W[16 * 256];   // 16x512 bf16
  __shared__ float o[16 * HD];
  __shared__ float mu[16], rsg[16];

  const int tid = threadIdx.x, lane = tid & 31, wv = tid >> 5;
  const int row0 = blockIdx.x * 16;
  __builtin_prefetch(Winp, 0, 0);

  for (int i = tid; i < 16 * 64; i += 128) {
    const int r = i >> 6, q = i & 63;
    const float2 f = *(const float2*)(hV1 + (size_t)(row0 + r) * HD + 2 * q);
    xW[i] = pack_bf16x2(f.x, f.y);
    xF[r * HD + 2 * q] = f.x;
    xF[r * HD + 2 * q + 1] = f.y;
  }
  __syncthreads();

  // layer 1: [16x128]x[128x512] -> GELU -> bf16
  v16bf a4[4];
#pragma unroll
  for (int kc = 0; kc < 4; ++kc) a4[kc] = load_a_frag(xW, 128, 0, kc, lane);
#pragma unroll
  for (int ntb = 0; ntb < 8; ++ntb) {
    const int nt = wv * 8 + ntb;
    v8f acc = {};
#pragma unroll
    for (int kc = 0; kc < 4; ++kc)
      acc = wmma_bf16(a4[kc], load_b_frag(Winp, 32, nt, kc, lane), acc);
    store_gelu_tile(t1W, 512, acc, 0, nt, bin, lane);
  }
  __syncthreads();

  // layer 2: [16x512]x[512x128] + bias + residual -> f32
#pragma unroll
  for (int t = 0; t < 2; ++t) {
    const int nt = 2 * wv + t;
    v8f acc = {};
#pragma unroll
    for (int kc = 0; kc < 16; ++kc)
      acc = wmma_bf16(load_a_frag(t1W, 512, 0, kc, lane),
                      load_b_frag(Woutp, 8, nt, kc, lane), acc);
    const int col = nt * 16 + (lane & 15);
    const float bv = bout[col];
#pragma unroll
    for (int i = 0; i < 8; ++i) {
      const int row = (lane >> 4) * 8 + i;
      o[row * HD + col] = acc[i] + bv + xF[row * HD + col];
    }
  }
  __syncthreads();

  if (tid < 16) {
    float s = 0.0f, s2 = 0.0f;
    for (int nn = 0; nn < HD; ++nn) { const float v = o[tid * HD + nn]; s += v; s2 += v * v; }
    const float mean = s * (1.0f / 128.0f);
    const float var  = s2 * (1.0f / 128.0f) - mean * mean;
    mu[tid] = mean; rsg[tid] = rsqrtf(var + 1e-5f);
  }
  __syncthreads();
  for (int i = tid; i < 16 * HD; i += 128) {
    const int r = i >> 7, nn = i & 127;
    const int node = row0 + r;
    float val = (o[i] - mu[r]) * rsg[r] * g2[nn] + be2[nn];
    val *= mask_V[node];
    hV2[(size_t)node * HD + nn] = val;
    out_hV[(size_t)node * HD + nn] = val;
  }
}

// ---------------- kernel 4: edge update + LN3 ----------------
__global__ __launch_bounds__(128) void enc_edge_kernel(
    const float* __restrict__ hV2, const float* __restrict__ h_E,
    const int* __restrict__ E_idx,
    const uint32_t* __restrict__ W1p, const float* __restrict__ b1,
    const uint32_t* __restrict__ W2p, const float* __restrict__ b2,
    const uint32_t* __restrict__ W3p, const float* __restrict__ b3,
    const float* __restrict__ g3, const float* __restrict__ be3,
    float* __restrict__ out_hE) {
  __shared__ uint32_t xW[32 * 192];
  __shared__ uint32_t y1W[32 * 64];
  __shared__ uint32_t y2W[32 * 64];
  __shared__ float m3[32 * HD];
  __shared__ float part[128], part2[128];
  __shared__ float mu[32], rs[32];
  __shared__ int idxSh[32];

  const int tid = threadIdx.x;
  const int node = blockIdx.x;
  const int b = node >> 11;
  __builtin_prefetch(W1p, 0, 0);

  stage_x(xW, hV2, h_E, E_idx, node, b, tid, idxSh);
  edge_mlp_32(xW, y1W, y2W, m3, W1p, b1, W2p, b2, W3p, b3, tid);

  // residual: h_E + me  (edge update is unmasked in reference)
  for (int i = tid; i < KNBR * HD; i += 128) {
    const int r = i >> 7, nn = i & 127;
    m3[r * HD + nn] += h_E[((size_t)node * KNBR + r) * HD + nn];
  }
  __syncthreads();

  // per-edge-row LayerNorm stats: 4 threads per row
  {
    const int r = tid >> 2, sub = tid & 3;
    float s = 0.0f, s2 = 0.0f;
    for (int nn = sub * 32; nn < sub * 32 + 32; ++nn) {
      const float v = m3[r * HD + nn];
      s += v; s2 += v * v;
    }
    part[tid] = s; part2[tid] = s2;
  }
  __syncthreads();
  if ((tid & 3) == 0) {
    const int r = tid >> 2;
    const float s  = part[tid] + part[tid + 1] + part[tid + 2] + part[tid + 3];
    const float s2 = part2[tid] + part2[tid + 1] + part2[tid + 2] + part2[tid + 3];
    const float mean = s * (1.0f / 128.0f);
    const float var  = s2 * (1.0f / 128.0f) - mean * mean;
    mu[r] = mean; rs[r] = rsqrtf(var + 1e-5f);
  }
  __syncthreads();
  for (int i = tid; i < KNBR * HD; i += 128) {
    const int r = i >> 7, nn = i & 127;
    out_hE[((size_t)node * KNBR + r) * HD + nn] =
        (m3[r * HD + nn] - mu[r]) * rs[r] * g3[nn] + be3[nn];
  }
}

// ---------------- host launcher ----------------
extern "C" void kernel_launch(void* const* d_in, const int* in_sizes, int n_in,
                              void* d_out, int out_size, void* d_ws, size_t ws_size,
                              hipStream_t stream) {
  (void)in_sizes; (void)n_in; (void)out_size; (void)ws_size;
  const float* h_V         = (const float*)d_in[0];
  const float* h_E         = (const float*)d_in[1];
  const int*   E_idx       = (const int*)d_in[2];
  const float* mask_V      = (const float*)d_in[3];
  const float* mask_attend = (const float*)d_in[4];
  // params in dict order
  const float* W1  = (const float*)d_in[5];  const float* b1  = (const float*)d_in[6];
  const float* W2  = (const float*)d_in[7];  const float* b2  = (const float*)d_in[8];
  const float* W3  = (const float*)d_in[9];  const float* b3  = (const float*)d_in[10];
  const float* W11 = (const float*)d_in[11]; const float* b11 = (const float*)d_in[12];
  const float* W12 = (const float*)d_in[13]; const float* b12 = (const float*)d_in[14];
  const float* W13 = (const float*)d_in[15]; const float* b13 = (const float*)d_in[16];
  const float* Win = (const float*)d_in[17]; const float* bin = (const float*)d_in[18];
  const float* Wout= (const float*)d_in[19]; const float* bout= (const float*)d_in[20];
  const float* g1  = (const float*)d_in[21]; const float* be1 = (const float*)d_in[22];
  const float* g2  = (const float*)d_in[23]; const float* be2 = (const float*)d_in[24];
  const float* g3  = (const float*)d_in[25]; const float* be3 = (const float*)d_in[26];

  uint32_t* Wp = (uint32_t*)d_ws;
  float* hV1 = (float*)(Wp + OFF_END);
  float* hV2 = hV1 + (size_t)BL * HD;
  float* out_hV = (float*)d_out;
  float* out_hE = out_hV + (size_t)BL * HD;

  auto pack = [&](const float* s, uint32_t* d, int KC, int NT, int N) {
    const int tot = KC * NT * 256;
    pack_w_kernel<<<(tot + 255) / 256, 256, 0, stream>>>(s, d, KC, NT, N);
  };
  pack(W1,  Wp + OFF_W1,  12,  8, 128);
  pack(W2,  Wp + OFF_W2,   4,  8, 128);
  pack(W3,  Wp + OFF_W3,   4,  8, 128);
  pack(W11, Wp + OFF_W11, 12,  8, 128);
  pack(W12, Wp + OFF_W12,  4,  8, 128);
  pack(W13, Wp + OFF_W13,  4,  8, 128);
  pack(Win, Wp + OFF_WIN,  4, 32, 512);
  pack(Wout,Wp + OFF_WOUT,16,  8, 128);

  enc_node_kernel<<<BL, 128, 0, stream>>>(h_V, h_E, E_idx, mask_attend,
                                          Wp + OFF_W1, b1, Wp + OFF_W2, b2, Wp + OFF_W3, b3,
                                          g1, be1, hV1);
  enc_ffn_kernel<<<BL / 16, 128, 0, stream>>>(hV1, Wp + OFF_WIN, bin, Wp + OFF_WOUT, bout,
                                              g2, be2, mask_V, hV2, out_hV);
  enc_edge_kernel<<<BL, 128, 0, stream>>>(hV2, h_E, E_idx,
                                          Wp + OFF_W11, b11, Wp + OFF_W12, b12, Wp + OFF_W13, b13,
                                          g3, be3, out_hE);
}
